// CausalSelfAttention_4483945857375
// MI455X (gfx1250) — matrix operand order repaired and checked by reading, not compiled, not run on hardware
//
#include <hip/hip_runtime.h>
#include <cstdint>
#include <cstddef>

#define DIM  2048
#define NH   16
#define HD   128
#define TSEQ 2048
#define BB   2

typedef __bf16 bf16;
typedef __attribute__((ext_vector_type(16))) __bf16 v16bf;
typedef __attribute__((ext_vector_type(8)))  __bf16 v8bf;
typedef __attribute__((ext_vector_type(8)))  float  v8f;

// ---------------------------------------------------------------------------
// f32 -> bf16 elementwise convert
// ---------------------------------------------------------------------------
__global__ void f32_to_bf16_kernel(const float* __restrict__ in,
                                   bf16* __restrict__ out, size_t n) {
  size_t i = (size_t)blockIdx.x * blockDim.x + threadIdx.x;
  if (i < n) out[i] = (bf16)in[i];
}

// ---------------------------------------------------------------------------
// C[M,N] (f32) = A[M,K] (bf16) * W[N,K]^T (bf16), via v_wmma_f32_16x16x32_bf16
// Block = 256 threads = 8 waves arranged 4(M) x 2(N); each wave computes a
// 32(M) x 64(N) tile (2 A-fragments x 4 B-fragments = 8 WMMA per k-step).
// Per k-step per lane: 64B A + 128B B for 8 WMMAs -> ~21 FLOP / L2 byte.
// Operand layouts (A and B share a contiguous k order):
//   A 16x32: lane (m=lane&15, half=lane>>4): elem e = K half*16+e
//            -> two contiguous 16B loads
//   B 32x16: lane (n=lane&15, half=lane>>4): elem e = K half*16+e
//            -> one contiguous 32B load along K of row n of W
//   C/D f32: elem r -> row half*8+r, col lane&15
// ---------------------------------------------------------------------------
__global__ __launch_bounds__(256) void gemm_bf16_wmma(
    const bf16* __restrict__ A, const bf16* __restrict__ W,
    float* __restrict__ C, int M, int N, int K) {
  const int lane = threadIdx.x & 31;
  const int wave = threadIdx.x >> 5;
  const int lm   = lane & 15;
  const int half = lane >> 4;
  const int mw   = wave & 3;   // 4 waves along M
  const int nw   = wave >> 2;  // 2 waves along N
  const int m0 = blockIdx.y * 128 + mw * 32;
  const int n0 = blockIdx.x * 128 + nw * 64;
  if (m0 >= M || n0 >= N) return;

  const bf16* arow0 = A + (size_t)(m0 + lm) * K;
  const bf16* arow1 = A + (size_t)(m0 + 16 + lm) * K;
  v8f acc[2][4] = {};

  for (int k0 = 0; k0 < K; k0 += 32) {
    const bf16* pa0 = arow0 + k0 + half * 16;
    const bf16* pa1 = arow1 + k0 + half * 16;
    v8bf a0lo = *(const v8bf*)pa0;
    v8bf a0hi = *(const v8bf*)(pa0 + 8);
    v8bf a1lo = *(const v8bf*)pa1;
    v8bf a1hi = *(const v8bf*)(pa1 + 8);
    v16bf af0, af1;
#pragma unroll
    for (int e = 0; e < 8; ++e) {
      af0[e] = a0lo[e]; af0[e + 8] = a0hi[e];
      af1[e] = a1lo[e]; af1[e + 8] = a1hi[e];
    }
    __builtin_prefetch(pa0 + 256, 0, 1);  // global_prefetch_b8, 8 k-steps ahead
    __builtin_prefetch(pa1 + 256, 0, 1);

    const bf16* wb = W + (size_t)(n0 + lm) * K + k0 + half * 16;
    v16bf b0 = *(const v16bf*)(wb);
    v16bf b1 = *(const v16bf*)(wb + 16 * (size_t)K);
    v16bf b2 = *(const v16bf*)(wb + 32 * (size_t)K);
    v16bf b3 = *(const v16bf*)(wb + 48 * (size_t)K);

    acc[0][0] = __builtin_amdgcn_wmma_f32_16x16x32_bf16(false, af0, false, b0, (short)0, acc[0][0], false, false);
    acc[0][1] = __builtin_amdgcn_wmma_f32_16x16x32_bf16(false, af0, false, b1, (short)0, acc[0][1], false, false);
    acc[0][2] = __builtin_amdgcn_wmma_f32_16x16x32_bf16(false, af0, false, b2, (short)0, acc[0][2], false, false);
    acc[0][3] = __builtin_amdgcn_wmma_f32_16x16x32_bf16(false, af0, false, b3, (short)0, acc[0][3], false, false);
    acc[1][0] = __builtin_amdgcn_wmma_f32_16x16x32_bf16(false, af1, false, b0, (short)0, acc[1][0], false, false);
    acc[1][1] = __builtin_amdgcn_wmma_f32_16x16x32_bf16(false, af1, false, b1, (short)0, acc[1][1], false, false);
    acc[1][2] = __builtin_amdgcn_wmma_f32_16x16x32_bf16(false, af1, false, b2, (short)0, acc[1][2], false, false);
    acc[1][3] = __builtin_amdgcn_wmma_f32_16x16x32_bf16(false, af1, false, b3, (short)0, acc[1][3], false, false);
  }

#pragma unroll
  for (int i = 0; i < 2; ++i)
#pragma unroll
    for (int r = 0; r < 8; ++r) {
      float* crow = C + (size_t)(m0 + i * 16 + half * 8 + r) * N + n0 + lm;
      crow[0]  = acc[i][0][r];
      crow[16] = acc[i][1][r];
      crow[32] = acc[i][2][r];
      crow[48] = acc[i][3][r];
    }
}

// ---------------------------------------------------------------------------
// RoPE + repack:  qkv f32 [b,t,3C]  ->  Q,K bf16 [b,h,t,d], V^T bf16 [b,h,d,t]
// One thread per (b,h,t,i), i in [0,64) handles the rotation pair (i, i+64).
// ---------------------------------------------------------------------------
__global__ void rope_pack_kernel(const float* __restrict__ qkv,
                                 bf16* __restrict__ Qb, bf16* __restrict__ Kb,
                                 bf16* __restrict__ Vt) {
  size_t tid = (size_t)blockIdx.x * blockDim.x + threadIdx.x;
  int    i = (int)(tid & 63);
  size_t t = (tid >> 6) & (TSEQ - 1);
  int    h = (int)((tid >> 17) & (NH - 1));
  int    b = (int)(tid >> 21);
  if (b >= BB) return;

  float inv = powf(10000.0f, -(float)(2 * i) / 128.0f);
  float ang = (float)t * inv;
  float c = cosf(ang), s = sinf(ang);

  const float* base = qkv + ((size_t)b * TSEQ + t) * (3 * DIM) + h * HD;
  float q1 = base[i],            q2 = base[i + 64];
  float k1 = base[DIM + i],      k2 = base[DIM + i + 64];
  float v1 = base[2 * DIM + i],  v2 = base[2 * DIM + i + 64];

  size_t bh = (size_t)b * NH + h;
  bf16* qo = Qb + (bh * TSEQ + t) * HD;
  bf16* ko = Kb + (bh * TSEQ + t) * HD;
  qo[i]      = (bf16)(q1 * c - q2 * s);
  qo[i + 64] = (bf16)(q1 * s + q2 * c);
  ko[i]      = (bf16)(k1 * c - k2 * s);
  ko[i + 64] = (bf16)(k1 * s + k2 * c);

  bf16* vo = Vt + bh * (size_t)HD * TSEQ;
  vo[(size_t)i * TSEQ + t]        = (bf16)v1;
  vo[(size_t)(i + 64) * TSEQ + t] = (bf16)v2;
}

// ---------------------------------------------------------------------------
// Flash attention, one wave per block. Wave owns a 16-row Q tile; streams
// 32-key tiles with online softmax; P transposed D-layout -> A-layout via LDS.
// Output written as bf16 [b,t,C] ready for the final WMMA GEMM.
// ---------------------------------------------------------------------------
__global__ __launch_bounds__(32) void flash_attn_kernel(
    const bf16* __restrict__ Qb, const bf16* __restrict__ Kb,
    const bf16* __restrict__ Vt, bf16* __restrict__ attn) {
  __shared__ bf16 lds[16 * 32];

  const int lane = threadIdx.x;
  const int lm   = lane & 15;
  const int half = lane >> 4;
  const int q0 = blockIdx.x * 16;
  const int h  = blockIdx.y;
  const int b  = blockIdx.z;

  const size_t bh = (size_t)b * NH + h;
  const bf16* Q  = Qb + bh * (size_t)TSEQ * HD;
  const bf16* Kp = Kb + bh * (size_t)TSEQ * HD;
  const bf16* Vp = Vt + bh * (size_t)HD * TSEQ;

  // Q fragments: 16 rows x 128 d as 4 A-fragments of 16x32
  v16bf qf[4];
  {
    const bf16* qrow = Q + (size_t)(q0 + lm) * HD;
#pragma unroll
    for (int j = 0; j < 4; ++j) {
      const bf16* p = qrow + j * 32 + half * 16;
      v8bf lo = *(const v8bf*)p;
      v8bf hi = *(const v8bf*)(p + 8);
#pragma unroll
      for (int e = 0; e < 8; ++e) { qf[j][e] = lo[e]; qf[j][e + 8] = hi[e]; }
    }
  }

  v8f o[8] = {};                         // 16 x 128 output accumulator
  float rowm[8], rowl[8];
#pragma unroll
  for (int r = 0; r < 8; ++r) { rowm[r] = -1e30f; rowl[r] = 0.0f; }
  const float scale = 0.08838834764831845f;  // 1/sqrt(128)

  for (int kv = 0; kv <= q0 + 15; kv += 32) {
    float sv[2][8];
    // ---- S = Q * K^T for two 16-key subtiles ----
#pragma unroll
    for (int t2 = 0; t2 < 2; ++t2) {
      v8f s = {};
      const int n  = kv + t2 * 16 + lm;
      const int nc = n < TSEQ ? n : TSEQ - 1;
      const bf16* krow = Kp + (size_t)nc * HD + half * 16;
#pragma unroll
      for (int j = 0; j < 4; ++j) {
        v16bf kf = *(const v16bf*)(krow + j * 32);
        s = __builtin_amdgcn_wmma_f32_16x16x32_bf16(false, qf[j], false, kf, (short)0, s, false, false);
      }
#pragma unroll
      for (int r = 0; r < 8; ++r) {
        int row = q0 + half * 8 + r;
        sv[t2][r] = (n <= row) ? s[r] * scale : -1e30f;  // causal mask
      }
    }

    // ---- online softmax: row max / exp / row sum (rows live in 16-lane halves)
    float nm[8], corr[8];
#pragma unroll
    for (int r = 0; r < 8; ++r) {
      float v = fmaxf(sv[0][r], sv[1][r]);
#pragma unroll
      for (int off = 8; off > 0; off >>= 1) v = fmaxf(v, __shfl_xor(v, off, 32));
      nm[r]   = fmaxf(rowm[r], v);
      corr[r] = __expf(rowm[r] - nm[r]);
      rowm[r] = nm[r];
    }
#pragma unroll
    for (int r = 0; r < 8; ++r) {
      float p0 = __expf(sv[0][r] - nm[r]);
      float p1 = __expf(sv[1][r] - nm[r]);
      sv[0][r] = p0; sv[1][r] = p1;
      float t = p0 + p1;
#pragma unroll
      for (int off = 8; off > 0; off >>= 1) t += __shfl_xor(t, off, 32);
      rowl[r] = rowl[r] * corr[r] + t;
    }
#pragma unroll
    for (int j = 0; j < 8; ++j)
#pragma unroll
      for (int r = 0; r < 8; ++r) o[j][r] *= corr[r];

    // ---- P: D-layout -> A-layout through LDS (16x32 bf16 row-major) ----
    __syncthreads();
#pragma unroll
    for (int t2 = 0; t2 < 2; ++t2)
#pragma unroll
      for (int r = 0; r < 8; ++r)
        lds[(half * 8 + r) * 32 + t2 * 16 + lm] = (bf16)sv[t2][r];
    __syncthreads();
    v16bf pf;
    {
      const bf16* p = &lds[lm * 32 + half * 16];
      v8bf lo = *(const v8bf*)p;
      v8bf hi = *(const v8bf*)(p + 8);
#pragma unroll
      for (int e = 0; e < 8; ++e) { pf[e] = lo[e]; pf[e + 8] = hi[e]; }
    }

    // ---- O += P * V  (V^T layout -> contiguous B-fragment loads along t) ----
    int kvc = kv + half * 16;
    if (kvc > TSEQ - 16) kvc = TSEQ - 16;   // masked tail, weights are 0
#pragma unroll
    for (int j = 0; j < 8; ++j) {
      v16bf vf = *(const v16bf*)(Vp + (size_t)(j * 16 + lm) * TSEQ + kvc);
      o[j] = __builtin_amdgcn_wmma_f32_16x16x32_bf16(false, pf, false, vf, (short)0, o[j], false, false);
    }
    __syncthreads();
  }

  // ---- epilogue: normalize and store bf16 [b,t,C] ----
#pragma unroll
  for (int r = 0; r < 8; ++r) {
    float inv = 1.0f / rowl[r];
    size_t row = (size_t)q0 + half * 8 + r;
    bf16* orow = attn + ((size_t)b * TSEQ + row) * DIM + h * HD + lm;
#pragma unroll
    for (int j = 0; j < 8; ++j) orow[j * 16] = (bf16)(o[j][r] * inv);
  }
}

// ---------------------------------------------------------------------------
extern "C" void kernel_launch(void* const* d_in, const int* in_sizes, int n_in,
                              void* d_out, int out_size, void* d_ws, size_t ws_size,
                              hipStream_t stream) {
  const float* x     = (const float*)d_in[0];
  const float* w_qkv = (const float*)d_in[1];
  const float* w_out = (const float*)d_in[2];
  float* out = (float*)d_out;

  char* ws = (char*)d_ws;
  size_t off = 0;
  auto take = [&](size_t bytes) -> char* {
    char* p = ws + off;
    off = (off + bytes + 255) & ~(size_t)255;
    return p;
  };

  const size_t nx    = (size_t)BB * TSEQ * DIM;   // 8.39M
  const size_t nwqkv = (size_t)3 * DIM * DIM;     // 12.6M
  const size_t nwout = (size_t)DIM * DIM;         // 4.19M

  bf16*  xb    = (bf16*)take(nx * 2);
  bf16*  wqkvb = (bf16*)take(nwqkv * 2);
  bf16*  woutb = (bf16*)take(nwout * 2);
  float* qkv   = (float*)take((size_t)BB * TSEQ * 3 * DIM * 4);
  bf16*  Qb    = (bf16*)take(nx * 2);
  bf16*  Kb    = (bf16*)take(nx * 2);
  bf16*  Vt    = (bf16*)take(nx * 2);
  bf16*  attnb = (bf16*)take(nx * 2);

  const int thr = 256;
  f32_to_bf16_kernel<<<(int)((nx    + thr - 1) / thr), thr, 0, stream>>>(x,     xb,    nx);
  f32_to_bf16_kernel<<<(int)((nwqkv + thr - 1) / thr), thr, 0, stream>>>(w_qkv, wqkvb, nwqkv);
  f32_to_bf16_kernel<<<(int)((nwout + thr - 1) / thr), thr, 0, stream>>>(w_out, woutb, nwout);

  // qkv = x @ w_qkv^T : M=4096, N=6144, K=2048 ; block tile 128x128
  dim3 g1(3 * DIM / 128, (BB * TSEQ) / 128);
  gemm_bf16_wmma<<<g1, 256, 0, stream>>>(xb, wqkvb, qkv, BB * TSEQ, 3 * DIM, DIM);

  size_t nrope = (size_t)BB * NH * TSEQ * 64;
  rope_pack_kernel<<<(int)((nrope + thr - 1) / thr), thr, 0, stream>>>(qkv, Qb, Kb, Vt);

  dim3 g2(TSEQ / 16, NH, BB);
  flash_attn_kernel<<<g2, 32, 0, stream>>>(Qb, Kb, Vt, attnb);

  // out = attn @ w_out^T : M=4096, N=2048, K=2048
  dim3 g3(DIM / 128, (BB * TSEQ) / 128);
  gemm_bf16_wmma<<<g3, 256, 0, stream>>>(attnb, woutb, out, BB * TSEQ, DIM, DIM);
}
